// ETGplusLSTM_43645457662515
// MI455X (gfx1250) — matrix operand order repaired and checked by reading, not compiled, not run on hardware
//
#include <hip/hip_runtime.h>
#include <hip/hip_fp16.h>

typedef __attribute__((ext_vector_type(16))) _Float16 v16h;
typedef __attribute__((ext_vector_type(8)))  float    v8f;

#define NBASIS 10
#define MAXR 5.0f

__device__ __forceinline__ float warp_sum32(float s) {
#pragma unroll
  for (int off = 16; off > 0; off >>= 1) s += __shfl_xor(s, off, 32);
  return s;
}

// ---------------- zero scratch ----------------
__global__ void zero_kernel(float* p, size_t n) {
  size_t i = (size_t)blockIdx.x * blockDim.x + threadIdx.x;
  size_t stride = (size_t)gridDim.x * blockDim.x;
  for (; i < n; i += stride) p[i] = 0.0f;
}

// ---------------- edge geometry: radial basis + cutoff ----------------
__global__ void edge_geom_kernel(const float* __restrict__ pos,
                                 const int* __restrict__ src,
                                 const int* __restrict__ dst,
                                 float* __restrict__ emb,
                                 float* __restrict__ cutoff, int E) {
  int e = blockIdx.x * blockDim.x + threadIdx.x;
  if (e >= E) return;
  int s = src[e], d = dst[e];
  float dx = pos[3*s+0] - pos[3*d+0];
  float dy = pos[3*s+1] - pos[3*d+1];
  float dz = pos[3*s+2] - pos[3*d+2];
  float r = sqrtf(dx*dx + dy*dy + dz*dz);
  const float width = MAXR / NBASIS;   // 0.5
#pragma unroll
  for (int b = 0; b < NBASIS; ++b) {
    float c = MAXR * (float)b / (float)(NBASIS - 1);   // linspace(0,5,10)
    float t = (r - c) / width;
    emb[(size_t)e * NBASIS + b] = __expf(-t * t);
  }
  float x = 10.0f * (1.0f - r / MAXR);
  cutoff[e] = (x > 0.0f) ? __expf(-1.0f / x) : 0.0f;  // soft_unit_step
}

// ---------------- pack activations into WMMA A-layout (f16, 16x32 tiles) ----
// Ap layout: [block][kc(2)][lane(32)][16 halves] ; row = 16*block + lane%16
__global__ void pack_a_kernel(const float* __restrict__ X, _Float16* __restrict__ Ap,
                              int N, int NB16) {
  int t = blockIdx.x * blockDim.x + threadIdx.x;  // one thread per (block,kc,lane)
  int total = NB16 * 2 * 32;
  if (t >= total) return;
  int lane = t & 31;
  int kc   = (t >> 5) & 1;
  int blk  = t >> 6;
  int row  = blk * 16 + (lane & 15);
  int g    = lane >> 4;
  _Float16* out = Ap + (size_t)t * 16;
#pragma unroll
  for (int j = 0; j < 16; ++j) {
    int v  = j >> 1;
    int kp = ((v < 4) ? (2 * v) : (16 + 2 * (v - 4))) + (j & 1) + 8 * g;
    int k  = kc * 32 + kp;
    float val = (row < N) ? X[(size_t)row * 64 + k] : 0.0f;
    out[j] = (_Float16)val;
  }
}

// ---------------- pack weights into WMMA B-layout (f16, 32x16 tiles) --------
// Wp layout: [m(8)][w(3)][kc(2)][nb(4)][lane(32)][16 halves]
__global__ void pack_w_kernel(const float* __restrict__ Wq,
                              const float* __restrict__ Wk,
                              const float* __restrict__ Wv,
                              _Float16* __restrict__ Wp) {
  int t = blockIdx.x * blockDim.x + threadIdx.x;  // per (m,w,kc,nb,lane)
  int total = 8 * 3 * 2 * 4 * 32;
  if (t >= total) return;
  int lane = t & 31;
  int nb   = (t >> 5) & 3;
  int kc   = (t >> 7) & 1;
  int w    = (t >> 8) % 3;
  int m    = (t >> 8) / 3;
  const float* W = (w == 0) ? Wq : ((w == 1) ? Wk : Wv);
  W += (size_t)m * 64 * 64;
  int g   = lane >> 4;
  int col = nb * 16 + (lane & 15);
  _Float16* out = Wp + (size_t)t * 16;
#pragma unroll
  for (int j = 0; j < 16; ++j) {
    int k = kc * 32 + 8 * g + ((j < 8) ? j : (8 + j));
    out[j] = (_Float16)W[(size_t)k * 64 + col];
  }
}

// ---------------- node projections via WMMA: Q,K,V = x @ W ------------------
// one wave per 16-row node block; 3 weights x 4 col-tiles x 2 k-steps = 24 wmma
__global__ void __launch_bounds__(256)
proj_gemm_kernel(const _Float16* __restrict__ Ap, const _Float16* __restrict__ Wp_mod,
                 float* __restrict__ Q, float* __restrict__ Kn, float* __restrict__ Vn,
                 int N, int NB16) {
  int wid  = (int)((blockIdx.x * blockDim.x + threadIdx.x) >> 5);
  int lane = threadIdx.x & 31;
  if (wid >= NB16) return;
  const v16h* Av = (const v16h*)Ap;
  const v16h* Bv = (const v16h*)Wp_mod;
  v16h a0 = Av[((size_t)wid * 2 + 0) * 32 + lane];
  v16h a1 = Av[((size_t)wid * 2 + 1) * 32 + lane];
  float* outs[3] = {Q, Kn, Vn};
  int g = lane >> 4;
  int colbase = lane & 15;
  int rowbase = wid * 16;
  bool full = (rowbase + 16 <= N);
#pragma unroll
  for (int w = 0; w < 3; ++w) {
    // one 64-bit base per output; all stores use immediate byte offsets
    float* base = outs[w] + (size_t)(rowbase + 8 * g) * 64 + colbase;
#pragma unroll
    for (int nb = 0; nb < 4; ++nb) {
      v16h b0 = Bv[(((size_t)w * 2 + 0) * 4 + nb) * 32 + lane];
      v16h b1 = Bv[(((size_t)w * 2 + 1) * 4 + nb) * 32 + lane];
      v8f acc = {};
      acc = __builtin_amdgcn_wmma_f32_16x16x32_f16(false, a0, false, b0,
                                                   (short)0, acc, false, false);
      acc = __builtin_amdgcn_wmma_f32_16x16x32_f16(false, a1, false, b1,
                                                   (short)0, acc, false, false);
      if (full) {
#pragma unroll
        for (int r = 0; r < 8; ++r) base[(size_t)r * 64 + nb * 16] = acc[r];
      } else {
#pragma unroll
        for (int r = 0; r < 8; ++r) {
          if (rowbase + r + 8 * g < N) base[(size_t)r * 64 + nb * 16] = acc[r];
        }
      }
    }
  }
}

// ---------------- edge pass 1: logits, exp, segment-sum z -------------------
// one wave32 per edge; lanes cover channels c and c+32
__global__ void __launch_bounds__(256)
edge_pass1_kernel(const float* __restrict__ Q, const float* __restrict__ Kn,
                  const float* __restrict__ Fk_m,
                  const int* __restrict__ src, const int* __restrict__ dst,
                  const float* __restrict__ emb, const float* __restrict__ cutoff,
                  float* __restrict__ exbuf, float* __restrict__ z, int E) {
  __shared__ float sF[NBASIS * 64];
  for (int i = threadIdx.x; i < NBASIS * 64; i += blockDim.x) sF[i] = Fk_m[i];
  __syncthreads();
  int lane = threadIdx.x & 31;
  int e = blockIdx.x * (blockDim.x >> 5) + (threadIdx.x >> 5);
  if (e >= E) return;
  int s = src[e], d = dst[e];
  float embv[NBASIS];
#pragma unroll
  for (int b = 0; b < NBASIS; ++b) embv[b] = emb[(size_t)e * NBASIS + b];
  float part = 0.0f;
#pragma unroll
  for (int h = 0; h < 2; ++h) {
    int c = lane + 32 * h;
    float ek = 0.0f;
#pragma unroll
    for (int b = 0; b < NBASIS; ++b) ek = fmaf(embv[b], sF[b * 64 + c], ek);
    float qv = Q[(size_t)d * 64 + c];
    float kv = Kn[(size_t)s * 64 + c] * ek;
    part = fmaf(qv, kv, part);
  }
  float tot = warp_sum32(part);
  float ex = cutoff[e] * __expf(tot * 0.125f);  // 1/sqrt(64)
  if (lane == 0) {
    exbuf[e] = ex;
    atomicAdd(&z[d], ex);
  }
}

// ---------------- edge pass 2: alpha, sqrt-weighted V scatter ---------------
__global__ void __launch_bounds__(256)
edge_pass2_kernel(const float* __restrict__ Vn, const float* __restrict__ Fv_m,
                  const int* __restrict__ src, const int* __restrict__ dst,
                  const float* __restrict__ emb, const float* __restrict__ exbuf,
                  const float* __restrict__ z, float* __restrict__ G, int E) {
  __shared__ float sF[NBASIS * 64];
  for (int i = threadIdx.x; i < NBASIS * 64; i += blockDim.x) sF[i] = Fv_m[i];
  __syncthreads();
  int lane = threadIdx.x & 31;
  int e = blockIdx.x * (blockDim.x >> 5) + (threadIdx.x >> 5);
  if (e >= E) return;
  int s = src[e], d = dst[e];
  float zv = z[d];
  zv = (zv == 0.0f) ? 1.0f : zv;
  float alpha = exbuf[e] / zv;
  float w = sqrtf(fmaxf(alpha, 0.0f));
  float embv[NBASIS];
#pragma unroll
  for (int b = 0; b < NBASIS; ++b) embv[b] = emb[(size_t)e * NBASIS + b];
#pragma unroll
  for (int h = 0; h < 2; ++h) {
    int c = lane + 32 * h;
    float ev = 0.0f;
#pragma unroll
    for (int b = 0; b < NBASIS; ++b) ev = fmaf(embv[b], sF[b * 64 + c], ev);
    float vv = Vn[(size_t)s * 64 + c] * ev;
    atomicAdd(&G[(size_t)d * 64 + c], w * vv);
  }
}

// ---------------- final LSTM-style gate combine -----------------------------
__global__ void combine_kernel(const float* __restrict__ G0, const float* __restrict__ G1,
                               const float* __restrict__ G2, const float* __restrict__ G3,
                               const float* __restrict__ C,
                               const float* __restrict__ wci, const float* __restrict__ bi,
                               const float* __restrict__ wcf, const float* __restrict__ bf_,
                               const float* __restrict__ bc,
                               const float* __restrict__ wco, const float* __restrict__ bo,
                               float* __restrict__ Hn, float* __restrict__ Cn, int total) {
  int i = blockIdx.x * blockDim.x + threadIdx.x;
  if (i >= total) return;
  int c = i & 63;
  float Cv = C[i];
  float I  = 1.0f / (1.0f + __expf(-(G0[i] + wci[c] * Cv + bi[c])));
  float F  = 1.0f / (1.0f + __expf(-(G1[i] + wcf[c] * Cv + bf_[c])));
  float Cnv = F * Cv + I * tanhf(G2[i] + bc[c]);
  float O  = 1.0f / (1.0f + __expf(-(G3[i] + wco[c] * Cnv + bo[c])));
  Hn[i] = O * tanhf(Cnv);
  Cn[i] = Cnv;
}

extern "C" void kernel_launch(void* const* d_in, const int* in_sizes, int n_in,
                              void* d_out, int out_size, void* d_ws, size_t ws_size,
                              hipStream_t stream) {
  const float* X   = (const float*)d_in[0];
  const float* pos = (const float*)d_in[1];
  const float* H   = (const float*)d_in[2];
  const float* C   = (const float*)d_in[3];
  const float* Wq  = (const float*)d_in[4];
  const float* Wk  = (const float*)d_in[5];
  const float* Wv  = (const float*)d_in[6];
  const float* Fk  = (const float*)d_in[7];
  const float* Fv  = (const float*)d_in[8];
  const float* wci = (const float*)d_in[9];
  const float* bi  = (const float*)d_in[10];
  const float* wcf = (const float*)d_in[11];
  const float* bf_ = (const float*)d_in[12];
  const float* bc  = (const float*)d_in[13];
  const float* wco = (const float*)d_in[14];
  const float* bo  = (const float*)d_in[15];
  const int* src   = (const int*)d_in[16];
  const int* dst   = (const int*)d_in[17];

  const int N = in_sizes[0] / 64;
  const int E = in_sizes[16];
  const int NB16 = (N + 15) / 16;

  // ---- carve workspace (256B aligned regions) ----
  char* p = (char*)d_ws;
  auto carve = [&](size_t bytes) -> char* {
    char* r = p;
    p += (bytes + 255) & ~(size_t)255;
    return r;
  };
  float*    emb    = (float*)carve((size_t)E * NBASIS * sizeof(float));
  float*    cutoff = (float*)carve((size_t)E * sizeof(float));
  float*    exbuf  = (float*)carve((size_t)E * sizeof(float));
  float*    zbuf   = (float*)carve((size_t)8 * N * sizeof(float));       // z per module
  float*    Gbuf   = (float*)carve((size_t)4 * N * 64 * sizeof(float));  // att pair sums
  float*    Qb     = (float*)carve((size_t)N * 64 * sizeof(float));
  float*    Knb    = (float*)carve((size_t)N * 64 * sizeof(float));
  float*    Vnb    = (float*)carve((size_t)N * 64 * sizeof(float));
  _Float16* ApX    = (_Float16*)carve((size_t)NB16 * 2 * 32 * 16 * sizeof(_Float16));
  _Float16* ApH    = (_Float16*)carve((size_t)NB16 * 2 * 32 * 16 * sizeof(_Float16));
  _Float16* Wp     = (_Float16*)carve((size_t)8 * 3 * 2 * 4 * 32 * 16 * sizeof(_Float16));
  (void)ws_size; (void)n_in; (void)out_size;

  // ---- zero accumulators (deterministic per call) ----
  zero_kernel<<<2048, 256, 0, stream>>>(zbuf, (size_t)8 * N);
  zero_kernel<<<2048, 256, 0, stream>>>(Gbuf, (size_t)4 * N * 64);

  // ---- precompute geometry + packed operands ----
  edge_geom_kernel<<<(E + 255) / 256, 256, 0, stream>>>(pos, src, dst, emb, cutoff, E);
  {
    int totA = NB16 * 2 * 32;
    pack_a_kernel<<<(totA + 255) / 256, 256, 0, stream>>>(X, ApX, N, NB16);
    pack_a_kernel<<<(totA + 255) / 256, 256, 0, stream>>>(H, ApH, N, NB16);
    int totW = 8 * 3 * 2 * 4 * 32;
    pack_w_kernel<<<(totW + 255) / 256, 256, 0, stream>>>(Wq, Wk, Wv, Wp);
  }

  const int gemmBlocks = (NB16 + 7) / 8;        // 8 waves per 256-thread block
  const int edgeBlocks = (E + 7) / 8;           // 1 wave per edge

  for (int m = 0; m < 8; ++m) {
    const _Float16* Ap = (m & 1) ? ApH : ApX;   // inputs = [X,H,X,H,X,H,X,H]
    const _Float16* Wpm = Wp + (size_t)m * 3 * 2 * 4 * 32 * 16;
    proj_gemm_kernel<<<gemmBlocks, 256, 0, stream>>>(Ap, Wpm, Qb, Knb, Vnb, N, NB16);
    edge_pass1_kernel<<<edgeBlocks, 256, 0, stream>>>(
        Qb, Knb, Fk + (size_t)m * NBASIS * 64, src, dst, emb, cutoff,
        exbuf, zbuf + (size_t)m * N, E);
    edge_pass2_kernel<<<edgeBlocks, 256, 0, stream>>>(
        Vnb, Fv + (size_t)m * NBASIS * 64, src, dst, emb, exbuf,
        zbuf + (size_t)m * N, Gbuf + (size_t)(m >> 1) * N * 64, E);
  }

  float* Hn = (float*)d_out;
  float* Cn = (float*)d_out + (size_t)N * 64;
  int total = N * 64;
  combine_kernel<<<(total + 255) / 256, 256, 0, stream>>>(
      Gbuf + 0 * (size_t)total, Gbuf + 1 * (size_t)total,
      Gbuf + 2 * (size_t)total, Gbuf + 3 * (size_t)total,
      C, wci, bi, wcf, bf_, bc, wco, bo, Hn, Cn, total);
}
